// MultiHeadAttention_58574763983773
// MI455X (gfx1250) — compile-verified, matrix operand order checked by reading, not run on hardware
//
#include <hip/hip_runtime.h>

// ---------- types ----------
typedef __bf16  bf16_t;
typedef bf16_t  v16bf __attribute__((ext_vector_type(16)));
typedef float   v8f   __attribute__((ext_vector_type(8)));
typedef int     v4i   __attribute__((ext_vector_type(4)));
typedef float   v4f   __attribute__((ext_vector_type(4)));
typedef unsigned int v2u __attribute__((ext_vector_type(2)));

#define LDS_STRIDE 72   // halves per row = 144 B: 16B-aligned, 36-dword stride -> 16 distinct start banks

// fp32 -> bf16, round-to-nearest-even
__device__ __forceinline__ unsigned short f2bf(float f) {
  unsigned int u = __float_as_uint(f);
  u += 0x7fffu + ((u >> 16) & 1u);
  return (unsigned short)(u >> 16);
}
__device__ __forceinline__ unsigned int f2bf2(float lo, float hi) {
  return (unsigned int)f2bf(lo) | ((unsigned int)f2bf(hi) << 16);
}

union FragU { v4i q[2]; v16bf v; };

// Load a 16x32 bf16 WMMA A/B fragment from an LDS tile stored row-major
// (row length >= kbase+32 halves, stride LDS_STRIDE halves).
// ISA layout: lanes 0-15 hold K {k0..k0+7, k0+16..k0+23}; lanes 16-31 the +8 halves.
__device__ __forceinline__ v16bf ld_frag(const unsigned short* base, int row, int kbase) {
  const int lane = threadIdx.x & 31;
  const unsigned short* p = base + row * LDS_STRIDE + kbase + ((lane >> 4) << 3);
  FragU f;
  f.q[0] = *reinterpret_cast<const v4i*>(p);        // ds_load_b128
  f.q[1] = *reinterpret_cast<const v4i*>(p + 16);   // ds_load_b128
  return f.v;
}

#define WMMA_BF16(a, b, c) \
  __builtin_amdgcn_wmma_f32_16x16x32_bf16(false, (a), false, (b), (short)0, (c), false, false)

// ---------------------------------------------------------------------------
// C[M,N] = A[M,K] @ W[N,K]^T + bias[N]      (fp32 in/out, bf16 WMMA math)
// Block: 256 thr (8 waves), tile 128x128x32. Wave (wm 0..3, wn 0..1) -> 32x64.
// ---------------------------------------------------------------------------
__global__ __launch_bounds__(256) void gemm_bias_kernel(
    const float* __restrict__ A, const float* __restrict__ W,
    const float* __restrict__ bias, float* __restrict__ C,
    int M, int N, int K)
{
  __shared__ unsigned short lA[128 * LDS_STRIDE];
  __shared__ unsigned short lB[128 * LDS_STRIDE];

  const int tid   = threadIdx.x;
  const int lane  = tid & 31;
  const int lan16 = lane & 15;
  const int half  = lane >> 4;
  const int wave  = tid >> 5;
  const int wm    = wave >> 1;          // 0..3
  const int wn    = wave & 1;           // 0..1
  const int m0    = blockIdx.y * 128;
  const int n0    = blockIdx.x * 128;

  v8f acc[2][4];
#pragma unroll
  for (int i = 0; i < 2; ++i)
#pragma unroll
    for (int j = 0; j < 4; ++j) acc[i][j] = {};

  for (int kb = 0; kb < K; kb += 32) {
    __syncthreads();
    // stage 128x32 of A and of W as bf16 (float4 global loads, b64 LDS stores)
#pragma unroll
    for (int it = 0; it < 4; ++it) {
      int idx = tid + it * 256;          // 0..1023
      int row = idx >> 3;                // 0..127
      int c4  = idx & 7;                 // col = c4*4
      v4f va = *reinterpret_cast<const v4f*>(A + (size_t)(m0 + row) * K + kb + c4 * 4);
      v4f vb = *reinterpret_cast<const v4f*>(W + (size_t)(n0 + row) * K + kb + c4 * 4);
      v2u pa; pa.x = f2bf2(va.x, va.y); pa.y = f2bf2(va.z, va.w);
      v2u pb; pb.x = f2bf2(vb.x, vb.y); pb.y = f2bf2(vb.z, vb.w);
      *reinterpret_cast<v2u*>(&lA[row * LDS_STRIDE + c4 * 4]) = pa;
      *reinterpret_cast<v2u*>(&lB[row * LDS_STRIDE + c4 * 4]) = pb;
    }
    __syncthreads();

    v16bf af[2], bfg[4];
#pragma unroll
    for (int mt = 0; mt < 2; ++mt) af[mt]  = ld_frag(lA, wm * 32 + mt * 16 + lan16, 0);
#pragma unroll
    for (int nt = 0; nt < 4; ++nt) bfg[nt] = ld_frag(lB, wn * 64 + nt * 16 + lan16, 0);
#pragma unroll
    for (int mt = 0; mt < 2; ++mt)
#pragma unroll
      for (int nt = 0; nt < 4; ++nt)
        acc[mt][nt] = WMMA_BF16(af[mt], bfg[nt], acc[mt][nt]);
  }

  // epilogue: bias add + fp32 store. C/D layout: VGPR j holds row j (+8 for hi half), lane = col.
#pragma unroll
  for (int nt = 0; nt < 4; ++nt) {
    int col = n0 + wn * 64 + nt * 16 + lan16;
    float bv = bias[col];
#pragma unroll
    for (int mt = 0; mt < 2; ++mt) {
      int mrow = m0 + wm * 32 + mt * 16 + half * 8;
#pragma unroll
      for (int j = 0; j < 8; ++j)
        C[(size_t)(mrow + j) * N + col] = acc[mt][nt][j] + bv;
    }
  }
}

// ---------------------------------------------------------------------------
// Flash attention forward. One block per (batch, head, 128 q rows).
// 8 waves, wave owns 16 q rows; KV streamed in 64-row tiles.
// ---------------------------------------------------------------------------
__global__ __launch_bounds__(256) void attn_kernel(
    const float* __restrict__ Qp, const float* __restrict__ Kp,
    const float* __restrict__ Vp, float* __restrict__ Op,
    int S, int D)
{
  __shared__ unsigned short lK[64 * LDS_STRIDE];    // K tile  [kv][d]
  __shared__ unsigned short lV[64 * LDS_STRIDE];    // V^T tile [d][kv]
  __shared__ unsigned short lP[128 * LDS_STRIDE];   // Q staging, then P (per-wave private rows)

  const int tid   = threadIdx.x;
  const int lane  = tid & 31;
  const int lan16 = lane & 15;
  const int half  = lane >> 4;
  const int wave  = tid >> 5;            // q rows [wave*16, wave*16+16)

  const int qt = blockIdx.x;
  const int h  = blockIdx.y;
  const int b  = blockIdx.z;

  const size_t bOff = (size_t)b * S * D;
  const int hcol = h * 64;
  const int q0   = qt * 128;

  // stage Q (pre-scaled by 1/sqrt(64)) into lP
#pragma unroll
  for (int it = 0; it < 8; ++it) {
    int idx = tid + it * 256;            // 0..2047
    int row = idx >> 4;                  // 0..127
    int c4  = idx & 15;                  // col = c4*4 in [0,64)
    v4f v = *reinterpret_cast<const v4f*>(Qp + bOff + (size_t)(q0 + row) * D + hcol + c4 * 4);
    v2u p; p.x = f2bf2(v.x * 0.125f, v.y * 0.125f); p.y = f2bf2(v.z * 0.125f, v.w * 0.125f);
    *reinterpret_cast<v2u*>(&lP[row * LDS_STRIDE + c4 * 4]) = p;
  }
  __syncthreads();

  v16bf qf0 = ld_frag(lP, wave * 16 + lan16, 0);
  v16bf qf1 = ld_frag(lP, wave * 16 + lan16, 32);

  v8f o[4];
#pragma unroll
  for (int i = 0; i < 4; ++i) o[i] = {};
  float mrow[8], lrow[8];
#pragma unroll
  for (int j = 0; j < 8; ++j) { mrow[j] = -1e30f; lrow[j] = 0.0f; }

  for (int kt = 0; kt < S / 64; ++kt) {
    __syncthreads();
    // stage K (row-major) and V (transposed) as bf16
#pragma unroll
    for (int it = 0; it < 4; ++it) {
      int idx = tid + it * 256;          // 0..1023
      int row = idx >> 4;                // 0..63 (kv index)
      int c4  = idx & 15;
      v4f k4 = *reinterpret_cast<const v4f*>(Kp + bOff + (size_t)(kt * 64 + row) * D + hcol + c4 * 4);
      v2u pk; pk.x = f2bf2(k4.x, k4.y); pk.y = f2bf2(k4.z, k4.w);
      *reinterpret_cast<v2u*>(&lK[row * LDS_STRIDE + c4 * 4]) = pk;
      v4f v4 = *reinterpret_cast<const v4f*>(Vp + bOff + (size_t)(kt * 64 + row) * D + hcol + c4 * 4);
      lV[(c4 * 4 + 0) * LDS_STRIDE + row] = f2bf(v4.x);
      lV[(c4 * 4 + 1) * LDS_STRIDE + row] = f2bf(v4.y);
      lV[(c4 * 4 + 2) * LDS_STRIDE + row] = f2bf(v4.z);
      lV[(c4 * 4 + 3) * LDS_STRIDE + row] = f2bf(v4.w);
    }
    __syncthreads();

    // S = (Q/8) K^T   -> 4 tiles of 16x16 per wave
    v8f sa[4];
#pragma unroll
    for (int nt = 0; nt < 4; ++nt) {
      sa[nt] = {};
      v16bf k0 = ld_frag(lK, nt * 16 + lan16, 0);
      v16bf k1 = ld_frag(lK, nt * 16 + lan16, 32);
      sa[nt] = WMMA_BF16(qf0, k0, sa[nt]);
      sa[nt] = WMMA_BF16(qf1, k1, sa[nt]);
    }

    // online softmax over this 64-wide KV slab
#pragma unroll
    for (int j = 0; j < 8; ++j) {
      float mc = fmaxf(fmaxf(sa[0][j], sa[1][j]), fmaxf(sa[2][j], sa[3][j]));
#pragma unroll
      for (int msk = 1; msk <= 8; msk <<= 1) mc = fmaxf(mc, __shfl_xor(mc, msk, 32));
      float mn   = fmaxf(mrow[j], mc);
      float corr = __expf(mrow[j] - mn);
      float rs = 0.0f;
#pragma unroll
      for (int nt = 0; nt < 4; ++nt) {
        float p = __expf(sa[nt][j] - mn);
        sa[nt][j] = p;
        rs += p;
      }
#pragma unroll
      for (int msk = 1; msk <= 8; msk <<= 1) rs += __shfl_xor(rs, msk, 32);
      lrow[j] = lrow[j] * corr + rs;
      mrow[j] = mn;
#pragma unroll
      for (int dt = 0; dt < 4; ++dt) o[dt][j] *= corr;
    }

    // spill P to this wave's private LDS rows (same-wave DS ops are in order -> no barrier)
#pragma unroll
    for (int nt = 0; nt < 4; ++nt)
#pragma unroll
      for (int j = 0; j < 8; ++j)
        lP[(wave * 16 + half * 8 + j) * LDS_STRIDE + nt * 16 + lan16] = f2bf(sa[nt][j]);

    v16bf pf0 = ld_frag(lP, wave * 16 + lan16, 0);
    v16bf pf1 = ld_frag(lP, wave * 16 + lan16, 32);
#pragma unroll
    for (int dt = 0; dt < 4; ++dt) {
      v16bf vt0 = ld_frag(lV, dt * 16 + lan16, 0);
      v16bf vt1 = ld_frag(lV, dt * 16 + lan16, 32);
      o[dt] = WMMA_BF16(pf0, vt0, o[dt]);
      o[dt] = WMMA_BF16(pf1, vt1, o[dt]);
    }
  }

  // normalize and store heads concatenated: out[b, s, h*64 + d]
#pragma unroll
  for (int dt = 0; dt < 4; ++dt) {
    int col = hcol + dt * 16 + lan16;
#pragma unroll
    for (int j = 0; j < 8; ++j) {
      int m = q0 + wave * 16 + half * 8 + j;
      Op[bOff + (size_t)m * D + col] = o[dt][j] / lrow[j];
    }
  }
}

// ---------------------------------------------------------------------------
extern "C" void kernel_launch(void* const* d_in, const int* in_sizes, int n_in,
                              void* d_out, int out_size, void* d_ws, size_t ws_size,
                              hipStream_t stream) {
  (void)in_sizes; (void)n_in; (void)out_size; (void)ws_size;

  const float* q   = (const float*)d_in[0];
  const float* k   = (const float*)d_in[1];
  const float* v   = (const float*)d_in[2];
  const float* w_q = (const float*)d_in[3];
  const float* b_q = (const float*)d_in[4];
  const float* w_k = (const float*)d_in[5];
  const float* b_k = (const float*)d_in[6];
  const float* w_v = (const float*)d_in[7];
  const float* b_v = (const float*)d_in[8];
  const float* w_o = (const float*)d_in[9];
  const float* b_o = (const float*)d_in[10];

  const int B = 2, S = 2048, D = 1024, H = 16;
  const int M = B * S;                       // 4096 tokens
  const size_t plane = (size_t)M * D;        // 4M floats = 16 MB

  float* qp = (float*)d_ws;                  // workspace: qp, kp, vp, attn_out (64 MB)
  float* kp = qp + plane;
  float* vp = kp + plane;
  float* ao = vp + plane;

  dim3 gblk(D / 128, M / 128);               // (8, 32)

  gemm_bias_kernel<<<gblk, 256, 0, stream>>>(q, w_q, b_q, qp, M, D, D);
  gemm_bias_kernel<<<gblk, 256, 0, stream>>>(k, w_k, b_k, kp, M, D, D);
  gemm_bias_kernel<<<gblk, 256, 0, stream>>>(v, w_v, b_v, vp, M, D, D);

  attn_kernel<<<dim3(S / 128, H, B), 256, 0, stream>>>(qp, kp, vp, ao, S, D);

  gemm_bias_kernel<<<gblk, 256, 0, stream>>>(ao, w_o, b_o, (float*)d_out, M, D, D);
}
